// Vector3DFeaturesWithIndividualFeature_50457275793730
// MI455X (gfx1250) — compile-verified, matrix operand order checked by reading, not run on hardware
//
#include <hip/hip_runtime.h>

typedef __attribute__((ext_vector_type(16))) _Float16 v16h;
typedef __attribute__((ext_vector_type(8)))  _Float16 v8h;
typedef __attribute__((ext_vector_type(8)))  float    v8f;

#define N_VOLS   24
#define N_PTS    32768
#define N_IND    18
#define N_OUT    144
#define FEAT_RES 16
#define N_FEAT   42
#define HIDDEN   32
#define D_IN     126     // N_FEAT * 3
#define DK       128     // K padded to multiple of 32
#define TP       128     // points per workgroup

#define SW1_STRIDE 136   // halfs per row of w1^T  (272B, 16B aligned, bank-staggered)
#define SW2_STRIDE 40    // halfs per row of w2^T  (80B, 16B aligned)
#define SH_STRIDE  40    // halfs per row of h     (80B, 16B aligned)

// Load a 16x32 (A) or 32x16 (B) f16 WMMA fragment from an LDS row.
// Element e of the v16h holds K = (e<8?0:16) + 8*laneHalf + (e&7), and runs
// of 8 elements are K-contiguous -> two 16B ds_load_b128 per fragment.
__device__ __forceinline__ v16h ld_frag(const _Float16* rowbase, int koff) {
  v8h lo = *(const v8h*)(rowbase + koff);
  v8h hi = *(const v8h*)(rowbase + koff + 16);
  v16h r;
#pragma unroll
  for (int i = 0; i < 8; ++i) { r[i] = lo[i]; r[i + 8] = hi[i]; }
  return r;
}

// Build the layer-1 A fragments in registers: this lane interpolates the 64
// channels (of its own point) that its lane-half owns in the A layout.
// HOFF = 8*laneHalf is a template arg so k (hence f=k/3, c=k%3) folds to
// compile-time constants.
template <int HOFF>
__device__ __forceinline__ void build_a(v16h (&a1)[4], const float* sFeatT,
                                        const int (&x0)[3], const float (&wgt)[3]) {
#pragma unroll
  for (int kb = 0; kb < 4; ++kb) {
#pragma unroll
    for (int e = 0; e < 16; ++e) {
      const int k = kb * 32 + (e < 8 ? 0 : 16) + HOFF + (e & 7);
      float val = 0.0f;
      if (k < D_IN) {
        const int f = k / 3;
        const int c = k - 3 * f;
        // sFeatT layout: [c][f][r]  (r contiguous)
        const int base = c * (N_FEAT * FEAT_RES) + f * FEAT_RES + x0[c];
        const float g0 = sFeatT[base];
        const float g1 = sFeatT[base + 1];
        val = fmaf(wgt[c], g1 - g0, g0);
      }
      a1[kb][e] = (_Float16)val;
    }
  }
}

__global__ __launch_bounds__(256) void vec3dfeat_mlp_kernel(
    const float* __restrict__ pts,         // (J, P, 3)
    const float* __restrict__ feature,     // (J, F, R, 3)
    const float* __restrict__ ind_feature, // (J, P, 18)
    const float* __restrict__ w1,          // (J, 126, 32)
    const float* __restrict__ b1,          // (J, 32)
    const float* __restrict__ w2,          // (J, 32, 126)
    const float* __restrict__ b2,          // (J, 126)
    float* __restrict__ out)               // (J, P, 144)
{
  __shared__ __align__(16) float    sFeatT[3 * N_FEAT * FEAT_RES]; // [c][f][r]
  __shared__ __align__(16) _Float16 sW1t[HIDDEN][SW1_STRIDE];      // [o][k]
  __shared__ __align__(16) _Float16 sW2t[DK][SW2_STRIDE];          // [n][k]
  __shared__ __align__(16) _Float16 sH[TP][SH_STRIDE];             // [p][k]
  __shared__ float sB1[HIDDEN];
  __shared__ float sB2[DK];

  const int tid = threadIdx.x;
  const int j   = blockIdx.x >> 8;          // volume
  const int pb  = blockIdx.x & 255;         // point-block
  const int p0  = pb * TP;

  const size_t volpt = (size_t)j * N_PTS + p0;

  // ---- independent-feature passthrough (pure global->global, overlaps) ----
  {
    const float* src = ind_feature + volpt * N_IND;
#pragma unroll 1
    for (int idx = tid; idx < TP * N_IND; idx += 256) {
      const int p = idx / N_IND;
      const int q = idx - p * N_IND;
      out[(volpt + p) * N_OUT + q] = src[idx];
    }
  }

  // ---- stage per-volume constants into LDS ----
  {
    const float* fsrc = feature + (size_t)j * (N_FEAT * FEAT_RES * 3);
#pragma unroll 1
    for (int idx = tid; idx < N_FEAT * FEAT_RES * 3; idx += 256) {
      const int fr = idx / 3;              // f*16 + r
      const int c  = idx - 3 * fr;
      sFeatT[c * (N_FEAT * FEAT_RES) + fr] = fsrc[idx];
    }
    const float* w1src = w1 + (size_t)j * (D_IN * HIDDEN);
#pragma unroll 1
    for (int idx = tid; idx < D_IN * HIDDEN; idx += 256) {
      const int i = idx >> 5;              // input channel (K)
      const int o = idx & 31;              // hidden (N)
      sW1t[o][i] = (_Float16)w1src[idx];
    }
    const float* w2src = w2 + (size_t)j * (HIDDEN * D_IN);
#pragma unroll 1
    for (int idx = tid; idx < HIDDEN * D_IN; idx += 256) {
      const int k = idx / D_IN;            // hidden (K)
      const int n = idx - k * D_IN;        // output channel (N)
      sW2t[n][k] = (_Float16)w2src[idx];
    }
    if (tid < HIDDEN) {                    // zero K-pad of w1^T (k = 126,127)
      sW1t[tid][126] = (_Float16)0.0f;
      sW1t[tid][127] = (_Float16)0.0f;
    }
    if (tid < 64) {                        // zero N-pad rows of w2^T
      sW2t[126 + (tid >> 5)][tid & 31] = (_Float16)0.0f;
    }
    if (tid < HIDDEN) sB1[tid] = b1[(size_t)j * HIDDEN + tid];
    if (tid < DK)     sB2[tid] = (tid < D_IN) ? b2[(size_t)j * D_IN + tid] : 0.0f;
  }

  // ---- per-lane grid-sample setup (each lane owns point m = wave*16 + lane%16)
  const int wv    = tid >> 5;    // wave id == M-tile
  const int lane  = tid & 31;
  const int lrow  = lane & 15;
  const int half_ = lane >> 4;
  const int myp   = wv * 16 + lrow;   // local point index

  int   x0[3];
  float wgt[3];
  {
    const float* pp = pts + (volpt + myp) * 3;
#pragma unroll
    for (int c = 0; c < 3; ++c) {
      float x = (pp[c] + 1.0f) * 0.5f * (float)(FEAT_RES - 1);
      x = fminf(fmaxf(x, 0.0f), (float)(FEAT_RES - 1));
      int i0 = (int)floorf(x);
      i0 = i0 > FEAT_RES - 2 ? FEAT_RES - 2 : (i0 < 0 ? 0 : i0);
      x0[c]  = i0;
      wgt[c] = x - (float)i0;
    }
  }

  __syncthreads();   // LDS staging complete

  // ---- layer 1: A fragments straight from interpolation (no LDS for V) ----
  v16h a1[4];
  if (half_) build_a<8>(a1, sFeatT, x0, wgt);
  else       build_a<0>(a1, sFeatT, x0, wgt);

#pragma unroll
  for (int nt = 0; nt < 2; ++nt) {
    const int ncol = nt * 16 + lrow;
    const float bias = sB1[ncol];
    v8f acc;
#pragma unroll
    for (int v = 0; v < 8; ++v) acc[v] = bias;
#pragma unroll
    for (int kb = 0; kb < 4; ++kb) {
      v16h bfrag = ld_frag(&sW1t[ncol][0], kb * 32 + 8 * half_);
      acc = __builtin_amdgcn_wmma_f32_16x16x32_f16(
          false, a1[kb], false, bfrag, (short)0, acc, false, false);
    }
    // ReLU + f16 convert, park in LDS (only this wave reads these rows back)
#pragma unroll
    for (int v = 0; v < 8; ++v) {
      const int m = wv * 16 + v + 8 * half_;
      sH[m][ncol] = (_Float16)fmaxf(acc[v], 0.0f);
    }
  }

  // ---- layer 2: o = relu(h) @ w2 + b2, direct global store (mask N-pad) ----
  v16h a2 = ld_frag(&sH[wv * 16 + lrow][0], 8 * half_);
#pragma unroll
  for (int nt = 0; nt < 8; ++nt) {
    const int col = nt * 16 + lrow;
    const float bias = sB2[col];
    v8f acc;
#pragma unroll
    for (int v = 0; v < 8; ++v) acc[v] = bias;
    v16h bfrag = ld_frag(&sW2t[col][0], 8 * half_);
    acc = __builtin_amdgcn_wmma_f32_16x16x32_f16(
        false, a2, false, bfrag, (short)0, acc, false, false);
    if (col < D_IN) {
#pragma unroll
      for (int v = 0; v < 8; ++v) {
        const int p = wv * 16 + v + 8 * half_;
        out[(volpt + p) * N_OUT + N_IND + col] = acc[v];
      }
    }
  }
}

extern "C" void kernel_launch(void* const* d_in, const int* in_sizes, int n_in,
                              void* d_out, int out_size, void* d_ws, size_t ws_size,
                              hipStream_t stream) {
  const float* pts         = (const float*)d_in[0];
  const float* feature     = (const float*)d_in[1];
  const float* ind_feature = (const float*)d_in[2];
  const float* w1          = (const float*)d_in[3];
  const float* b1          = (const float*)d_in[4];
  const float* w2          = (const float*)d_in[5];
  const float* b2          = (const float*)d_in[6];
  float* out = (float*)d_out;

  dim3 grid(N_VOLS * (N_PTS / TP));   // 24 * 256 = 6144 blocks
  dim3 block(256);                    // 8 wave32
  vec3dfeat_mlp_kernel<<<grid, block, 0, stream>>>(
      pts, feature, ind_feature, w1, b1, w2, b2, out);
}
